// KroneckerMarkerCovariance_45311904972911
// MI455X (gfx1250) — matrix operand order repaired and checked by reading, not compiled, not run on hardware
//
#include <hip/hip_runtime.h>

// Problem dims (match reference)
#define GRID_N 128
#define CDIM   32
#define MDIM   64

#define SB   136   // K2: bf16 row stride (128+8): 272B rows, 16B aligned, bank-spread
#define SB2  40    // K1/K3: 32-elem K rows + pad (80B, 16B aligned)
#define JST  648   // K1/K3: per-j stride in bf16 (16*SB2 + 8), keeps bank spread

typedef __attribute__((ext_vector_type(16))) __bf16 v16bf;
typedef __attribute__((ext_vector_type(8)))  __bf16 v8bf;
typedef __attribute__((ext_vector_type(8)))  float  v8f;

__device__ __forceinline__ v8f vzero() {
  v8f z;
#pragma unroll
  for (int q = 0; q < 8; ++q) z[q] = 0.f;
  return z;
}

// ---- vectorized fragment loads (16B LDS b128 accesses) ---------------------
// B-fragment: lane's 16 K-values are contiguous (layout [N][K], base += kh*16)
__device__ __forceinline__ v16bf ldfrag16(const __bf16 *p) {
  p = (const __bf16 *)__builtin_assume_aligned(p, 16);
  v16bf r;
  __builtin_memcpy(&r, p, 32);
  return r;
}
// A-fragment: two contiguous 8-element K runs of lane's row (layout [M][K])
__device__ __forceinline__ v16bf ldfragA(const __bf16 *p, int kh) {
  const __bf16 *p0 = (const __bf16 *)__builtin_assume_aligned(p + kh * 16, 16);
  const __bf16 *p1 = (const __bf16 *)__builtin_assume_aligned(p + kh * 16 + 8, 16);
  v16bf r;
  __builtin_memcpy(reinterpret_cast<char *>(&r), p0, 16);
  __builtin_memcpy(reinterpret_cast<char *>(&r) + 16, p1, 16);
  return r;
}

// split-once producer store: 8 consecutive rows at one column -> [col][row]
__device__ __forceinline__ void store8_split(__bf16 *__restrict__ hi,
                                             __bf16 *__restrict__ lo,
                                             int off, const v8f &a) {
  v8bf h8, l8;
#pragma unroll
  for (int r = 0; r < 8; ++r) {
    float x = a[r];
    __bf16 h = (__bf16)x;
    h8[r] = h;
    l8[r] = (__bf16)(x - (float)h);
  }
  __bf16 *ph = (__bf16 *)__builtin_assume_aligned(hi + off, 16);
  __bf16 *pl = (__bf16 *)__builtin_assume_aligned(lo + off, 16);
  __builtin_memcpy(ph, &h8, 16);
  __builtin_memcpy(pl, &l8, 16);
}

// elementwise A builder (tiny V_C matrix only; built once per kernel)
template <typename F>
__device__ __forceinline__ void build_a(int lane, F get, v16bf &hi, v16bf &lo) {
  const int M = lane & 15, kh = lane >> 4;
#pragma unroll
  for (int t = 0; t < 8; ++t) {
#pragma unroll
    for (int b = 0; b < 2; ++b) {
      const int K = kh * 16 + 2 * t + b;
      float x = get(M, K);
      __bf16 h = (__bf16)x;
      hi[2 * t + b] = h;
      lo[2 * t + b] = (__bf16)(x - (float)h);
    }
  }
}

// fp32-accurate product via 3 bf16 WMMAs (drop lo*lo term)
__device__ __forceinline__ v8f wmma3(const v16bf &ah, const v16bf &al,
                                     const v16bf &bh, const v16bf &bl, v8f c) {
  c = __builtin_amdgcn_wmma_f32_16x16x32_bf16(false, ah, false, bh, (short)0, c, false, false);
  c = __builtin_amdgcn_wmma_f32_16x16x32_bf16(false, ah, false, bl, (short)0, c, false, false);
  c = __builtin_amdgcn_wmma_f32_16x16x32_bf16(false, al, false, bh, (short)0, c, false, false);
  return c;
}

// ---------------------------------------------------------------------------
// 128x128x128 GEMM on split-bf16 LDS arrays. A: [M][K] stride SB. B: [N][K]
// stride SB (i.e. B transposed). B fragments are software double-buffered one
// tile ahead so ds_load_b128s issue before the WMMAs that hide them.
// Epilogue gets (rowbase, col, acc8): lane owns rows rowbase..+7 of col.
// ---------------------------------------------------------------------------
template <typename EPI>
__device__ __forceinline__ void gemm128s(int wave, int lane,
                                         const __bf16 *__restrict__ Ah,
                                         const __bf16 *__restrict__ Al,
                                         const __bf16 *__restrict__ Bh,
                                         const __bf16 *__restrict__ Bl,
                                         EPI epi) {
  const int ln = lane & 15, kh = lane >> 4;
  v8f acc[8];
#pragma unroll
  for (int nt = 0; nt < 8; ++nt) acc[nt] = vzero();

#pragma unroll
  for (int ks = 0; ks < 4; ++ks) {
    const int abase = (wave * 16 + ln) * SB + ks * 32;
    v16bf ah = ldfragA(Ah + abase, kh);
    v16bf al = ldfragA(Al + abase, kh);
    const int bstep = ln * SB + ks * 32 + kh * 16;
    v16bf bh = ldfrag16(Bh + bstep);
    v16bf bl = ldfrag16(Bl + bstep);
#pragma unroll
    for (int nt = 0; nt < 8; ++nt) {
      v16bf nbh = bh, nbl = bl;
      if (nt < 7) {
        const int nb = (nt + 1) * 16 * SB + bstep;
        nbh = ldfrag16(Bh + nb);
        nbl = ldfrag16(Bl + nb);
      }
      acc[nt] = wmma3(ah, al, bh, bl, acc[nt]);
      bh = nbh;
      bl = nbl;
    }
  }
  const int r0 = wave * 16 + kh * 8;
#pragma unroll
  for (int nt = 0; nt < 8; ++nt) epi(r0, nt * 16 + ln, acc[nt]);
}

// ---------------------------------------------------------------------------
// Kernel 1: marker forward.  Per (i,j): D[k,m] = sum_c V_C[c,k] * X[c,m].
// Slab staged as split bf16 [j][m][c] (B^T layout), output transposed to
// [k][m][i][j] through an f32 LDS stage (128B-run global writes).
// ---------------------------------------------------------------------------
__global__ void k_marker_fwd(const float *__restrict__ v,
                             const float *__restrict__ VC,
                             float *__restrict__ tmp) {
  extern __shared__ char smraw[];
  __bf16 *Xh = (__bf16 *)smraw;          // [32 j][16 m][SB2 c]
  __bf16 *Xl = Xh + 32 * JST;
  float  *So = (float *)(Xl + 32 * JST); // [32 k][16 m][33 j]
  const int tid = threadIdx.x, lane = tid & 31, wave = tid >> 5;
  const int i = blockIdx.x >> 2, jb = blockIdx.x & 3;
  const int ln = lane & 15, kh = lane >> 4;

  v16bf a_hi[2], a_lo[2];
#pragma unroll
  for (int mt = 0; mt < 2; ++mt)   // A = V_C^T
    build_a(lane, [&](int M, int K) { return VC[K * CDIM + mt * 16 + M]; },
            a_hi[mt], a_lo[mt]);

  for (int nt = 0; nt < 4; ++nt) {
    const int m0 = nt * 16;
    // stage + split: Xh/Xl[j][m][c]  (global reads: 64B runs over m)
    for (int idx = tid; idx < 32 * 32 * 16; idx += 256) {
      const int jl = idx >> 9;
      const int c  = (idx >> 4) & 31;
      const int mm = idx & 15;
      float x = v[(size_t)(i * GRID_N + jb * 32 + jl) * (CDIM * MDIM) + c * MDIM + m0 + mm];
      __bf16 h = (__bf16)x;
      Xh[jl * JST + mm * SB2 + c] = h;
      Xl[jl * JST + mm * SB2 + c] = (__bf16)(x - (float)h);
    }
    __syncthreads();
#pragma unroll
    for (int jj = 0; jj < 4; ++jj) {
      const int jl = wave * 4 + jj;
      const int bb = jl * JST + ln * SB2 + kh * 16;
      v16bf bh = ldfrag16(Xh + bb);
      v16bf bl = ldfrag16(Xl + bb);
#pragma unroll
      for (int mt = 0; mt < 2; ++mt) {
        v8f acc = wmma3(a_hi[mt], a_lo[mt], bh, bl, vzero());
#pragma unroll
        for (int r = 0; r < 8; ++r) {
          const int k = mt * 16 + kh * 8 + r;
          So[(k * 16 + ln) * 33 + jl] = acc[r];
        }
      }
    }
    __syncthreads();
    // coalesced 128B-run writes: tmp[k][m0+mm][i][jb*32+jl]
    for (int idx = tid; idx < 32 * 16 * 32; idx += 256) {
      const int k  = idx >> 9;
      const int mm = (idx >> 5) & 15;
      const int jl = idx & 31;
      tmp[((size_t)(k * MDIM + m0 + mm) * GRID_N + i) * GRID_N + jb * 32 + jl] =
          So[(k * 16 + mm) * 33 + jl];
    }
    __syncthreads();
  }
}

// ---------------------------------------------------------------------------
// Kernel 2: fused spatial transform, in place on tmp (layout [k][m][i][j]).
// T' = V * ((V^T T V) ./ lambda[:,:,k]) * V^T, via:
//   stage W0=Lay(T^T), Lay(V), Lay(V^T)  (split once)
//   G1: D1 = T^T*V          -> emit Lay(D1^T)
//   G2: D2 = V^T*D1 = S^T   -> .*rcp(lambda), emit Lay(S)
//   G3: D3 = S*V^T          -> emit Lay(D3^T)
//   G4: D4 = V*D3 = T'      -> global (coalesced)
// All operands vector-loaded; all LDS epilogues are b128 stores.
// ---------------------------------------------------------------------------
__global__ void k_spatial_mid(const float *__restrict__ Vg,
                              const float *__restrict__ te,
                              float *__restrict__ tmp) {
  extern __shared__ char smraw[];
  __bf16 *Ah0 = (__bf16 *)smraw;      // buffer A (hi/lo)
  __bf16 *Al0 = Ah0 + 128 * SB;
  __bf16 *Bh0 = Al0 + 128 * SB;       // buffer B (hi/lo)
  __bf16 *Bl0 = Bh0 + 128 * SB;
  __bf16 *Vh  = Bl0 + 128 * SB;       // Lay(V)
  __bf16 *Vl  = Vh + 128 * SB;
  __bf16 *Vth = Vl + 128 * SB;        // Lay(V^T)
  __bf16 *Vtl = Vth + 128 * SB;

  const int tid = threadIdx.x, lane = tid & 31, wave = tid >> 5;
  const int slab = blockIdx.x;        // = k*64 + m
  const int kidx = slab >> 6;
  float *G = tmp + (size_t)slab * (GRID_N * GRID_N);

  for (int idx = tid; idx < GRID_N * GRID_N; idx += 256) {
    const int r = idx >> 7, c = idx & 127;
    float vv = Vg[idx];
    __bf16 vh = (__bf16)vv, vl = (__bf16)(vv - (float)vh);
    Vh[r * SB + c]  = vh;  Vl[r * SB + c]  = vl;   // Lay(V)
    Vth[c * SB + r] = vh;  Vtl[c * SB + r] = vl;   // Lay(V^T)
    float tv = G[idx];
    __bf16 th = (__bf16)tv;
    Ah0[c * SB + r] = th;                           // W0 = Lay(T^T)
    Al0[c * SB + r] = (__bf16)(tv - (float)th);
  }
  __syncthreads();

  // G1: D1 = T^T * V
  gemm128s(wave, lane, Ah0, Al0, Vth, Vtl,
           [&](int r0, int col, const v8f &a) {
             store8_split(Bh0, Bl0, col * SB + r0, a);
           });
  __syncthreads();
  // G2: D2 = V^T * D1 = S^T ; scale by fast reciprocal of lambda ; emit Lay(S)
  gemm128s(wave, lane, Vth, Vtl, Bh0, Bl0,
           [&](int r0, int col, v8f a) {
#pragma unroll
             for (int r = 0; r < 8; ++r)
               a[r] *= __builtin_amdgcn_rcpf(
                   te[(size_t)(col * GRID_N + r0 + r) * CDIM + kidx]);
             store8_split(Ah0, Al0, col * SB + r0, a);
           });
  __syncthreads();
  // G3: D3 = S * V^T
  gemm128s(wave, lane, Ah0, Al0, Vh, Vl,
           [&](int r0, int col, const v8f &a) {
             store8_split(Bh0, Bl0, col * SB + r0, a);
           });
  __syncthreads();
  // G4: T' = V * D3  -> global, 64B-run stores, in place
  gemm128s(wave, lane, Vh, Vl, Bh0, Bl0,
           [&](int r0, int col, const v8f &a) {
#pragma unroll
             for (int r = 0; r < 8; ++r) G[(r0 + r) * GRID_N + col] = a[r];
           });
}

// ---------------------------------------------------------------------------
// Kernel 3: marker reverse.  out[i,j,c,m] = sum_k V_C[c,k] * Z[k,m,i,j].
// Staged split bf16 [j][m][k] from coalesced 128B-run reads of [k][m][i][j];
// next m-chunk's lines are prefetched (global_prefetch) during compute.
// ---------------------------------------------------------------------------
__global__ void k_marker_rev(const float *__restrict__ tmp,
                             const float *__restrict__ VC,
                             float *__restrict__ out) {
  extern __shared__ char smraw[];
  __bf16 *Zh = (__bf16 *)smraw;       // [32 j][16 m][SB2 k]
  __bf16 *Zl = Zh + 32 * JST;
  const int tid = threadIdx.x, lane = tid & 31, wave = tid >> 5;
  const int i = blockIdx.x >> 2, jb = blockIdx.x & 3;
  const int ln = lane & 15, kh = lane >> 4;

  v16bf a_hi[2], a_lo[2];
#pragma unroll
  for (int mt = 0; mt < 2; ++mt)   // A = V_C
    build_a(lane, [&](int M, int K) { return VC[(mt * 16 + M) * CDIM + K]; },
            a_hi[mt], a_lo[mt]);

  for (int nt = 0; nt < 4; ++nt) {
    const int m0 = nt * 16;
    for (int idx = tid; idx < 32 * 16 * 32; idx += 256) {
      const int k  = idx >> 9;
      const int mm = (idx >> 5) & 15;
      const int jl = idx & 31;
      float x = tmp[((size_t)(k * MDIM + m0 + mm) * GRID_N + i) * GRID_N + jb * 32 + jl];
      __bf16 h = (__bf16)x;
      Zh[jl * JST + mm * SB2 + k] = h;
      Zl[jl * JST + mm * SB2 + k] = (__bf16)(x - (float)h);
    }
    // prefetch next chunk's 512 cachelines while this chunk computes
    if (nt < 3) {
      for (int idx = tid; idx < 32 * 16; idx += 256) {
        const int k  = idx >> 4;
        const int mm = idx & 15;
        __builtin_prefetch(
            &tmp[((size_t)(k * MDIM + m0 + 16 + mm) * GRID_N + i) * GRID_N + jb * 32],
            0, 1);
      }
    }
    __syncthreads();
#pragma unroll
    for (int jj = 0; jj < 4; ++jj) {
      const int jl = wave * 4 + jj;
      const int j  = jb * 32 + jl;
      float *Y = out + (size_t)(i * GRID_N + j) * (CDIM * MDIM);
      const int bb = jl * JST + ln * SB2 + kh * 16;
      v16bf bh = ldfrag16(Zh + bb);
      v16bf bl = ldfrag16(Zl + bb);
#pragma unroll
      for (int mt = 0; mt < 2; ++mt) {
        v8f acc = wmma3(a_hi[mt], a_lo[mt], bh, bl, vzero());
#pragma unroll
        for (int r = 0; r < 8; ++r) {
          const int c = mt * 16 + kh * 8 + r;
          Y[c * MDIM + m0 + ln] = acc[r];
        }
      }
    }
    __syncthreads();
  }
}

// ---------------------------------------------------------------------------
// launch: v, V_C, V, triple_eigs -> out ; d_ws holds one 128 MB intermediate
// ---------------------------------------------------------------------------
extern "C" void kernel_launch(void *const *d_in, const int *in_sizes, int n_in,
                              void *d_out, int out_size, void *d_ws, size_t ws_size,
                              hipStream_t stream) {
  const float *v  = (const float *)d_in[0];
  const float *VC = (const float *)d_in[1];
  const float *Vg = (const float *)d_in[2];
  const float *te = (const float *)d_in[3];
  float *out = (float *)d_out;
  float *tmp = (float *)d_ws; // n*n*C*m floats = 128 MB

  const int smem_k1 = 2 * 32 * JST * 2 + 32 * 16 * 33 * 4; // 150,528 B
  const int smem_k2 = 8 * 128 * SB * 2;                    // 278,528 B (<320KB)
  const int smem_k3 = 2 * 32 * JST * 2;                    //  82,944 B

  hipLaunchKernelGGL(k_marker_fwd, dim3(GRID_N * 4), dim3(256), smem_k1, stream,
                     v, VC, tmp);
  hipLaunchKernelGGL(k_spatial_mid, dim3(CDIM * MDIM), dim3(256), smem_k2, stream,
                     Vg, te, tmp);
  hipLaunchKernelGGL(k_marker_rev, dim3(GRID_N * 4), dim3(256), smem_k3, stream,
                     tmp, VC, out);
}
